// RGCN_10634339025195
// MI455X (gfx1250) — matrix operand order repaired and checked by reading, not compiled, not run on hardware
//
#include <hip/hip_runtime.h>
#include <hip/hip_bf16.h>
#include <math.h>

// ---------------------------------------------------------------------------
// RGCN: B=512, N=128, F=128, K=4.
// One workgroup per batch element b; all 3 layers computed in LDS.
//   stage:      E[b] f16 panel (128KB) -> LDS via GLOBAL_LOAD_ASYNC_TO_LDS_B128
//   per layer:  G_k = H @ RW_k  (4 GEMMs 128x128x128) -> scatter to LDS as
//               G2 column-major [(n) x (j*4+k)]  (512x128 f16, 128KB)
//               msg = E_flat[128x512] @ G2[512x128]   (all operands in LDS)
//               Hs  = H @ SW
//               H'  = sigmoid(msg * D[b,i] + Hs)
// GEMMs via v_wmma_f32_16x16x32_f16 (f16 in, f32 accumulate).
// LDS: 32KB (H) + 128KB (G2) + 128KB (E) = 288KB (needs CDNA5 320KB WGP LDS).
// ---------------------------------------------------------------------------

typedef _Float16 v16h __attribute__((ext_vector_type(16)));
typedef _Float16 v8h  __attribute__((ext_vector_type(8)));
typedef float    v8f  __attribute__((ext_vector_type(8)));

#define Bsz 512
#define Nn  128
#define Ff  128
#define Kk  4

__device__ __forceinline__ v16h cat8(v8h a, v8h b) {
  return __builtin_shufflevector(a, b, 0,1,2,3,4,5,6,7,8,9,10,11,12,13,14,15);
}

// A operand: 16x32 f16 tile from row-major matrix (ld halves), tile at (row0,k0).
// Lanes 0-15 -> M=lane, K: VGPR0-3 = k0+0..7, VGPR4-7 = k0+16..23;
// lanes 16-31 -> M=lane-16, K = k0+8..15 and k0+24..31.
__device__ __forceinline__ v16h load_a16x32(const _Float16* base, int ld,
                                            int row0, int k0, int lane) {
  int m  = row0 + (lane & 15);
  int hi = (lane >> 4) & 1;
  const _Float16* p = base + m * ld + k0 + 8 * hi;
  v8h lo = *(const v8h*)p;
  v8h hv = *(const v8h*)(p + 16);
  return cat8(lo, hv);
}

// B operand: 32x16 f16 tile from COLUMN-major matrix Bcm[n][k] (ld = k-stride
// in halves), tile at (k0, n0). Per lane: k0+8*hi+0..7 and k0+16+8*hi+0..7.
__device__ __forceinline__ v16h load_b32x16cm(const _Float16* base, int ld,
                                              int k0, int n0, int lane) {
  int n  = n0 + (lane & 15);
  int hi = (lane >> 4) & 1;
  const _Float16* p = base + n * ld + k0 + 8 * hi;
  v8h lo = *(const v8h*)p;
  v8h hv = *(const v8h*)(p + 16);
  return cat8(lo, hv);
}

__device__ __forceinline__ v8f wmma_f16(v16h a, v16h b, v8f c) {
  return __builtin_amdgcn_wmma_f32_16x16x32_f16(false, a, false, b,
                                                (short)0, c, false, false);
}

__device__ __forceinline__ float sigmoidf_fast(float x) {
  return 1.0f / (1.0f + __expf(-x));
}

// --------------------------- prologue kernels ------------------------------

__global__ void rgcn_convert_e(const float* __restrict__ E,
                               _Float16* __restrict__ Ef, int n) {
  int stride = gridDim.x * blockDim.x;
  for (int i = blockIdx.x * blockDim.x + threadIdx.x; i < n; i += stride)
    Ef[i] = (_Float16)E[i];
}

// D[b,i] = 1 / (sum_{j,k} E[b,i,j,k] + K). One wave32 per row of 512 floats.
__global__ void rgcn_compute_d(const float* __restrict__ E,
                               float* __restrict__ D) {
  int row  = (blockIdx.x * blockDim.x + threadIdx.x) >> 5;  // b*128 + i
  int lane = threadIdx.x & 31;
  const float* e = E + (size_t)row * (Nn * Kk);
  float s = 0.0f;
  for (int t = lane; t < Nn * Kk; t += 32) s += e[t];
  #pragma unroll
  for (int off = 16; off > 0; off >>= 1) s += __shfl_xor(s, off, 32);
  if (lane == 0) D[row] = 1.0f / (s + (float)Kk);
}

// Column-major f16 weights:
//   RWt[l][k][n][m] = RW_l[m, n, k];  SWt[l][n][m] = SW_l[m, n]
__global__ void rgcn_convert_w(const float* __restrict__ RW1,
                               const float* __restrict__ RW2,
                               const float* __restrict__ RW3,
                               const float* __restrict__ SW1,
                               const float* __restrict__ SW2,
                               const float* __restrict__ SW3,
                               _Float16* __restrict__ RWt,
                               _Float16* __restrict__ SWt) {
  const int nrw = 3 * Kk * Ff * Ff;          // 196608
  const int nsw = 3 * Ff * Ff;               // 49152
  int i = blockIdx.x * blockDim.x + threadIdx.x;
  if (i < nrw) {
    int l = i / (Kk * Ff * Ff);
    int r = i % (Kk * Ff * Ff);
    int k = r >> 14;
    int n = (r >> 7) & 127;
    int m = r & 127;
    const float* src = (l == 0) ? RW1 : (l == 1) ? RW2 : RW3;
    RWt[i] = (_Float16)src[m * (Ff * Kk) + n * Kk + k];
  } else if (i < nrw + nsw) {
    int j = i - nrw;
    int l = j / (Ff * Ff);
    int r = j % (Ff * Ff);
    int n = r >> 7;
    int m = r & 127;
    const float* src = (l == 0) ? SW1 : (l == 1) ? SW2 : SW3;
    SWt[j] = (_Float16)src[m * Ff + n];
  }
}

// ----------------------------- main kernel ---------------------------------
// grid = 512 workgroups (one per b), 256 threads = 8 wave32.
// dynamic LDS: sH 32KB | sG 128KB | sE 128KB = 288KB.
__global__ void __launch_bounds__(256)
rgcn_layers(const float* __restrict__ Hin, const _Float16* __restrict__ Ef,
            const float* __restrict__ Dws, const _Float16* __restrict__ RWt,
            const _Float16* __restrict__ SWt, float* __restrict__ Hout) {
  extern __shared__ __align__(64) char smem[];
  _Float16* sH = (_Float16*)smem;                        // [128][128] row-major
  _Float16* sG = (_Float16*)(smem + Nn * Ff * 2);        // [n=128][jk=512] cm
  _Float16* sE = (_Float16*)(smem + Nn * Ff * 2 + Nn * Nn * Kk * 2);  // [128][512]

  const int b    = blockIdx.x;
  const int tid  = threadIdx.x;
  const int lane = tid & 31;
  const int wave = tid >> 5;           // 0..7, owns output rows 16w..16w+15
  const int hi   = (lane >> 4) & 1;
  const int nl   = lane & 15;

  // ---- async-stage E[b] f16 panel into LDS (16B per lane per issue) ----
  {
    const _Float16* gsrc = Ef + (size_t)b * (Nn * Nn * Kk);
    unsigned ldsBase = (unsigned)(size_t)(const void*)sE;  // low 32b = LDS off
    for (int idx = tid; idx < (Nn * Nn * Kk) / 8; idx += 256) {
      unsigned loff = ldsBase + (unsigned)idx * 16u;
      unsigned long long ga = (unsigned long long)(size_t)(gsrc + idx * 8);
      asm volatile("global_load_async_to_lds_b128 %0, %1, off"
                   :: "v"(loff), "v"(ga) : "memory");
    }
  }

  // stage H[b] fp32 -> f16 LDS (overlaps with async E copy)
  const float* Hb = Hin + (size_t)b * Nn * Ff;
  for (int idx = tid; idx < Nn * Ff; idx += 256) sH[idx] = (_Float16)Hb[idx];

  asm volatile("s_wait_asynccnt 0" ::: "memory");
  __syncthreads();

  float* HoutB = Hout + (size_t)b * Nn * Ff;

  for (int layer = 0; layer < 3; ++layer) {
    const _Float16* rwt = RWt + layer * (Kk * Ff * Ff);
    const _Float16* swt = SWt + layer * (Ff * Ff);
    if (layer < 2)  // warm L2/WGP$ for next layer's weights
      __builtin_prefetch(RWt + (layer + 1) * (Kk * Ff * Ff), 0, 0);

    // ---- phase 1: G_k = H @ RW_k, scattered to sG[n*512 + j*4 + k] ----
    // 256 output tiles (4k x 8jt x 8nt), 32 per wave, 4 WMMA k-steps each.
    #pragma unroll 1
    for (int t = 0; t < 32; ++t) {
      int id = wave * 32 + t;
      int k  = id >> 6;
      int jt = (id >> 3) & 7;
      int nt = id & 7;
      const _Float16* bk = rwt + k * (Ff * Ff);
      v8f g = {};
      #pragma unroll
      for (int s = 0; s < 4; ++s)
        g = wmma_f16(load_a16x32(sH, Ff, jt * 16, s * 32, lane),
                     load_b32x16cm(bk, Ff, s * 32, nt * 16, lane), g);
      int n = nt * 16 + nl;
      #pragma unroll
      for (int e = 0; e < 8; ++e) {
        int j = jt * 16 + e + 8 * hi;
        sG[n * (Nn * Kk) + j * Kk + k] = (_Float16)g[e];
      }
    }
    __syncthreads();  // sG ready; all waves done reading sH in phase 1

    // ---- phase 2: msg = E_flat @ G2 (K-dim 512), Hs = H @ SW ----
    v8f acc[8] = {};
    #pragma unroll 2
    for (int s = 0; s < 16; ++s) {
      v16h a = load_a16x32(sE, Nn * Kk, wave * 16, s * 32, lane);
      #pragma unroll
      for (int nt = 0; nt < 8; ++nt)
        acc[nt] = wmma_f16(a, load_b32x16cm(sG, Nn * Kk, s * 32, nt * 16, lane),
                           acc[nt]);
    }
    v8f hacc[8] = {};
    #pragma unroll
    for (int s = 0; s < 4; ++s) {
      v16h a = load_a16x32(sH, Ff, wave * 16, s * 32, lane);
      #pragma unroll
      for (int nt = 0; nt < 8; ++nt)
        hacc[nt] = wmma_f16(a, load_b32x16cm(swt, Ff, s * 32, nt * 16, lane),
                            hacc[nt]);
    }

    // ---- combine: H' = sigmoid(msg * D + Hs); wave writes only its strip ----
    float dv[8];
    #pragma unroll
    for (int e = 0; e < 8; ++e)
      dv[e] = Dws[b * Nn + wave * 16 + e + 8 * hi];
    #pragma unroll
    for (int nt = 0; nt < 8; ++nt) {
      int n = nt * 16 + nl;
      #pragma unroll
      for (int e = 0; e < 8; ++e) {
        int i = wave * 16 + e + 8 * hi;
        float v = sigmoidf_fast(acc[nt][e] * dv[e] + hacc[nt][e]);
        if (layer == 2) HoutB[i * Ff + n] = v;       // final layer -> global
        else            sH[i * Ff + n] = (_Float16)v;
      }
    }
    __syncthreads();  // new H visible before next layer's phase 1
  }
}

// ------------------------------- MLP head ----------------------------------
__global__ void rgcn_head_partial(const float* __restrict__ Hout,
                                  const float* __restrict__ W1,
                                  const float* __restrict__ b1,
                                  const float* __restrict__ W2,
                                  const float* __restrict__ b2,
                                  float* __restrict__ partial) {
  __shared__ float red[256];
  int row = blockIdx.x * blockDim.x + threadIdx.x;  // 0..65535
  const float* h = Hout + (size_t)row * Ff;
  float z[20];
  #pragma unroll
  for (int j = 0; j < 20; ++j) z[j] = b1[j];
  for (int m = 0; m < Ff; ++m) {
    float hv = h[m];
    #pragma unroll
    for (int j = 0; j < 20; ++j) z[j] += hv * W1[m * 20 + j];
  }
  float y = b2[0];
  #pragma unroll
  for (int j = 0; j < 20; ++j) {
    float a = z[j];
    a = (a > 0.0f) ? a : 0.1f * a;
    y += a * W2[j];
  }
  red[threadIdx.x] = sigmoidf_fast(y);
  __syncthreads();
  for (int off = 128; off > 0; off >>= 1) {
    if (threadIdx.x < off) red[threadIdx.x] += red[threadIdx.x + off];
    __syncthreads();
  }
  if (threadIdx.x == 0) partial[blockIdx.x] = red[0];
}

__global__ void rgcn_head_final(const float* __restrict__ partial,
                                float* __restrict__ out) {
  __shared__ float red[256];
  red[threadIdx.x] = partial[threadIdx.x];
  __syncthreads();
  for (int off = 128; off > 0; off >>= 1) {
    if (threadIdx.x < off) red[threadIdx.x] += red[threadIdx.x + off];
    __syncthreads();
  }
  if (threadIdx.x == 0) out[0] = red[0] * (1.0f / 65536.0f);
}

// ------------------------------- launcher ----------------------------------
extern "C" void kernel_launch(void* const* d_in, const int* in_sizes, int n_in,
                              void* d_out, int out_size, void* d_ws,
                              size_t ws_size, hipStream_t stream) {
  const float* H   = (const float*)d_in[0];
  const float* E   = (const float*)d_in[1];
  const float* RW1 = (const float*)d_in[2];
  const float* SW1 = (const float*)d_in[3];
  const float* RW2 = (const float*)d_in[4];
  const float* SW2 = (const float*)d_in[5];
  const float* RW3 = (const float*)d_in[6];
  const float* SW3 = (const float*)d_in[7];
  const float* W1  = (const float*)d_in[8];
  const float* b1  = (const float*)d_in[9];
  const float* W2  = (const float*)d_in[10];
  const float* b2  = (const float*)d_in[11];
  float* out = (float*)d_out;

  // workspace carve-up (bytes)
  char* ws = (char*)d_ws;
  const size_t nE = (size_t)Bsz * Nn * Nn * Kk;         // 33,554,432
  _Float16* Ef   = (_Float16*)ws;                       // 67,108,864 B
  size_t off     = nE * 2;
  float*   Hout  = (float*)(ws + off);  off += (size_t)Bsz * Nn * Ff * 4;
  float*   Dws   = (float*)(ws + off);  off += (size_t)Bsz * Nn * 4;
  _Float16* RWt  = (_Float16*)(ws + off); off += (size_t)3 * Kk * Ff * Ff * 2;
  _Float16* SWt  = (_Float16*)(ws + off); off += (size_t)3 * Ff * Ff * 2;
  float*   part  = (float*)(ws + off);  off += 256 * 4;
  (void)ws_size; (void)in_sizes; (void)n_in; (void)out_size;

  rgcn_convert_e<<<8192, 256, 0, stream>>>(E, Ef, (int)nE);
  rgcn_compute_d<<<(Bsz * Nn) / 8, 256, 0, stream>>>(E, Dws);
  rgcn_convert_w<<<(3 * Kk * Ff * Ff + 3 * Ff * Ff + 255) / 256, 256, 0,
                   stream>>>(RW1, RW2, RW3, SW1, SW2, SW3, RWt, SWt);

  const int ldsBytes = (Nn * Ff + Nn * Nn * Kk + Nn * Nn * Kk) * 2;  // 288KB
  (void)hipFuncSetAttribute((const void*)rgcn_layers,
                            hipFuncAttributeMaxDynamicSharedMemorySize,
                            ldsBytes);
  rgcn_layers<<<Bsz, 256, ldsBytes, stream>>>(H, Ef, Dws, RWt, SWt, Hout);

  rgcn_head_partial<<<256, 256, 0, stream>>>(Hout, W1, b1, W2, b2, part);
  rgcn_head_final<<<1, 256, 0, stream>>>(part, out);
}
